// Sage14FX_31009663877208
// MI455X (gfx1250) — matrix operand order repaired and checked by reading, not compiled, not run on hardware
//
#include <hip/hip_runtime.h>
#include <math.h>

typedef __bf16 bf16;
typedef __attribute__((ext_vector_type(16))) __bf16 v16bf;
typedef __attribute__((ext_vector_type(8)))  float  v8f;

// ---------------------------------------------------------------- utilities
__global__ void k_zero(float* p, int n){
  int i = blockIdx.x * 256 + threadIdx.x;
  if (i < n) p[i] = 0.f;
}

__global__ void k_f2bf(const float* __restrict__ s, bf16* __restrict__ d, int n){
  int i = blockIdx.x * 256 + threadIdx.x;
  if (i < n) d[i] = (bf16)s[i];
}

// transpose-convert: src f32 [K][N] -> dst bf16 [N][K]
__global__ void k_f2bf_T(const float* __restrict__ s, bf16* __restrict__ d, int K, int N){
  int i = blockIdx.x * 256 + threadIdx.x;
  if (i >= N * K) return;
  int n = i / K, k = i % K;
  d[i] = (bf16)s[(size_t)k * N + n];
}

// conv weights: src f32 [9][ci 256][co 256] -> dst bf16 [9][co 256][ci 256]
__global__ void k_wT(const float* __restrict__ s, bf16* __restrict__ d){
  int i = blockIdx.x * 256 + threadIdx.x;
  if (i >= 589824) return;
  int kk = i >> 16; int r = i & 65535; int co = r >> 8; int ci = r & 255;
  d[i] = (bf16)s[(size_t)(kk * 256 + ci) * 256 + co];
}

// ---------------------------------------------------------------- conv1 (direct, VALU; only ~13 GFLOP)
__global__ __launch_bounds__(256)
void k_conv1(const float* __restrict__ x, const float* __restrict__ w,
             const float* __restrict__ b, bf16* __restrict__ out){
  int pid = blockIdx.x;            // img*1024 + y*32 + x
  int img = pid >> 10;
  int py  = (pid >> 5) & 31;
  int px  = pid & 31;
  int co  = threadIdx.x;
  __shared__ float patch[99];
  int t = threadIdx.x;
  if (t < 99){
    int ci = t % 11; int kk = t / 11; int kx = kk % 3; int ky = kk / 3;
    int iy = py + ky - 1, ix = px + kx - 1;
    float v = 0.f;
    if (iy >= 0 && iy < 32 && ix >= 0 && ix < 32)
      v = x[(((size_t)img * 32 + iy) * 32 + ix) * 11 + ci];
    patch[t] = v;
  }
  __syncthreads();
  float acc = b[co];
  for (int k = 0; k < 99; k++) acc = fmaf(patch[k], w[k * 256 + co], acc);
  acc = fmaxf(acc, 0.f);
  out[(size_t)pid * 256 + co] = (bf16)acc;
}

// ---------------------------------------------------------------- generic bf16 WMMA GEMM
// C[M,N] = act( A[M,K] * B + bias ), A bf16 row-major [M,K], BT bf16 TRANSPOSED [N,K].
// block = 128 thr = 4 waves; tile 64(M) x 64(N); wave w owns rows [16w,16w+16), all 4 n-tiles.
__global__ __launch_bounds__(128)
void k_gemm_bf16(const bf16* __restrict__ A, const bf16* __restrict__ BT,
                 const float* __restrict__ bias, float* __restrict__ C,
                 int M, int N, int K, int relu){
  __shared__ __align__(16) bf16 As[64][40];   // 80B rows: 16B-aligned, 20-dword stride (5 x 16B)
  __shared__ __align__(16) bf16 Bt[64][40];   // [n][k] fragment-friendly
  int tid  = threadIdx.x;
  int lane = tid & 31;
  int wv   = tid >> 5;
  int m0 = blockIdx.y * 64;
  int n0 = blockIdx.x * 64;
  v8f acc[4];
  #pragma unroll
  for (int i = 0; i < 4; i++){
    #pragma unroll
    for (int r = 0; r < 8; r++) acc[i][r] = 0.f;
  }
  const uint4 zz = make_uint4(0u, 0u, 0u, 0u);
  for (int k0 = 0; k0 < K; k0 += 32){
    { // stage A 64x32 (rows = M): row-uniform guard, 2x b128 load + 2x b128 LDS store
      int r = tid >> 1, c0 = (tid & 1) * 16;
      int gm = m0 + r;
      uint4 d0 = zz, d1 = zz;
      if (gm < M){
        const uint4* src = (const uint4*)(A + (size_t)gm * K + k0 + c0);
        d0 = src[0]; d1 = src[1];
      }
      uint4* dst = (uint4*)(&As[r][c0]);
      dst[0] = d0; dst[1] = d1;
    }
    { // stage B 64x32 (rows = N, transposed weights: contiguous in K)
      int r = tid >> 1, c0 = (tid & 1) * 16;
      int gn = n0 + r;
      uint4 d0 = zz, d1 = zz;
      if (gn < N){
        const uint4* src = (const uint4*)(BT + (size_t)gn * K + k0 + c0);
        d0 = src[0]; d1 = src[1];
      }
      uint4* dst = (uint4*)(&Bt[r][c0]);
      dst[0] = d0; dst[1] = d1;
    }
    if (k0 + 32 < K)
      __builtin_prefetch(A + (size_t)(m0 + (tid >> 1)) * K + k0 + 32, 0, 0);
    __syncthreads();
    // A frag: lane L -> row M=L%16; K = (e>>3)*16 + (L>>4)*8 + (e&7)  (two 16B runs)
    v16bf a;
    int mr = wv * 16 + (lane & 15);
    int kh = (lane >> 4) << 3;
    #pragma unroll
    for (int e = 0; e < 8; e++){
      a[e]     = As[mr][kh + e];
      a[e + 8] = As[mr][16 + kh + e];
    }
    // B frag: lane L -> col N=L%16; K = (e>>3)*16 + (L>>4)*8 + (e&7)  (two 16B runs)
    int kb = (lane >> 4) << 4;
    #pragma unroll
    for (int nt = 0; nt < 4; nt++){
      v16bf bq;
      int nc = nt * 16 + (lane & 15);
      #pragma unroll
      for (int e = 0; e < 8; e++){ bq[e] = Bt[nc][kh + e]; bq[e + 8] = Bt[nc][16 + kh + e]; }
      acc[nt] = __builtin_amdgcn_wmma_f32_16x16x32_bf16(
          false, a, false, bq, (short)0, acc[nt], false, false);
    }
    __syncthreads();
  }
  int rbase = (lane >> 4) * 8;
  int cl = lane & 15;
  #pragma unroll
  for (int nt = 0; nt < 4; nt++){
    int col = n0 + nt * 16 + cl;
    float bv = (bias && col < N) ? bias[col] : 0.f;
    #pragma unroll
    for (int r = 0; r < 8; r++){
      int row = m0 + wv * 16 + rbase + r;
      if (row < M && col < N){
        float v = acc[nt][r] + bv;
        if (relu) v = fmaxf(v, 0.f);
        C[(size_t)row * N + col] = v;
      }
    }
  }
}

// ---------------------------------------------------------------- implicit-GEMM 3x3 conv (bf16 WMMA)
// in bf16 [nimg][H][W][256], wT bf16 [3][3][co 256][ci 256], relu, out bf16 [nimg][H][W][256]
__global__ __launch_bounds__(256)
void k_conv3_wmma(const bf16* __restrict__ in, const bf16* __restrict__ wT,
                  const float* __restrict__ bias, bf16* __restrict__ out,
                  int H, int W){
  int blk = blockIdx.x;
  int img = blk / H;
  int y   = blk % H;
  int tid  = threadIdx.x;
  int lane = tid & 31;
  int wv   = tid >> 5;
  int mt   = wv & 1;
  int ntb  = (wv >> 1) * 4;
  __shared__ __align__(16) bf16 Arow[3][34][32];   // padded input rows (x = -1 .. W)
  __shared__ __align__(16) bf16 Wc[256][40];       // weight chunk [co][ci 32] (transposed global)
  v8f acc[4];
  #pragma unroll
  for (int i = 0; i < 4; i++){
    #pragma unroll
    for (int r = 0; r < 8; r++) acc[i][r] = 0.f;
  }
  const size_t imgbase = (size_t)img * H * W * 256;
  const uint4 zz = make_uint4(0u, 0u, 0u, 0u);
  for (int c0 = 0; c0 < 256; c0 += 32){
    // stage 3 padded rows: one thread per (ky,xx) position, 64B vector copy or zero
    int tot = 3 * (W + 2);
    for (int i = tid; i < tot; i += 256){
      int xx = i % (W + 2);
      int ky = i / (W + 2);
      int iy = y + ky - 1;
      int ix = xx - 1;
      uint4 d0 = zz, d1 = zz, d2 = zz, d3 = zz;
      if (iy >= 0 && iy < H && ix >= 0 && ix < W){
        const uint4* src = (const uint4*)(in + imgbase + ((size_t)iy * W + ix) * 256 + c0);
        d0 = src[0]; d1 = src[1]; d2 = src[2]; d3 = src[3];
      }
      uint4* dst = (uint4*)(&Arow[ky][xx][0]);
      dst[0] = d0; dst[1] = d1; dst[2] = d2; dst[3] = d3;
    }
    __syncthreads();
    for (int ky = 0; ky < 3; ky++){
      for (int kx = 0; kx < 3; kx++){
        __syncthreads();                       // retire previous Wc consumers
        { // stage: thread t owns co=t, 32 contiguous ci (64B) from transposed weights
          int co = tid;
          const uint4* src = (const uint4*)(wT + ((size_t)(ky * 3 + kx) * 256 + co) * 256 + c0);
          uint4* dst = (uint4*)(&Wc[co][0]);
          dst[0] = src[0]; dst[1] = src[1]; dst[2] = src[2]; dst[3] = src[3];
        }
        __syncthreads();
        v16bf a;
        int px = mt * 16 + (lane & 15);
        int xi = px + kx; if (xi > W + 1) xi = 0;   // dead pixels, masked at store
        int kh = (lane >> 4) << 3;
        #pragma unroll
        for (int e = 0; e < 8; e++){
          a[e]     = Arow[ky][xi][kh + e];
          a[e + 8] = Arow[ky][xi][16 + kh + e];
        }
        int kb = (lane >> 4) << 4;
        #pragma unroll
        for (int nt = 0; nt < 4; nt++){
          v16bf bq;
          int nc = (ntb + nt) * 16 + (lane & 15);
          #pragma unroll
          for (int e = 0; e < 8; e++){ bq[e] = Wc[nc][kh + e]; bq[e + 8] = Wc[nc][16 + kh + e]; }
          acc[nt] = __builtin_amdgcn_wmma_f32_16x16x32_bf16(
              false, a, false, bq, (short)0, acc[nt], false, false);
        }
      }
    }
    __syncthreads();                            // retire Arow consumers
  }
  int rbase = (lane >> 4) * 8;
  int cl = lane & 15;
  #pragma unroll
  for (int nt = 0; nt < 4; nt++){
    int co = (ntb + nt) * 16 + cl;
    float bv = bias[co];
    #pragma unroll
    for (int r = 0; r < 8; r++){
      int px = mt * 16 + rbase + r;
      if (px < W){
        float v = fmaxf(acc[nt][r] + bv, 0.f);
        out[imgbase + ((size_t)y * W + px) * 256 + co] = (bf16)v;
      }
    }
  }
}

// ---------------------------------------------------------------- LayerNorm(ch) + spatial-mean accumulate
__global__ __launch_bounds__(256)
void k_ln_mean(const bf16* __restrict__ act2, const float* __restrict__ g,
               const float* __restrict__ b, float* __restrict__ feats){
  int blk = blockIdx.x;              // img*4 blocks
  int img = blk >> 2;
  int pix = (blk & 3) * 256 + threadIdx.x;
  const bf16* row = act2 + ((size_t)img * 1024 + pix) * 256;
  float s = 0.f, s2 = 0.f;
  for (int c = 0; c < 256; c++){ float v = (float)row[c]; s += v; s2 += v * v; }
  float mu = s * (1.f / 256.f);
  float var = fmaxf(s2 * (1.f / 256.f) - mu * mu, 0.f);
  float rs = rsqrtf(var + 1e-3f);
  __shared__ float accum[256];
  accum[threadIdx.x] = 0.f;
  __syncthreads();
  for (int i = 0; i < 256; i++){
    int c = (threadIdx.x + i) & 255;
    float v = ((float)row[c] - mu) * rs * g[c] + b[c];
    atomicAdd(&accum[c], v);
  }
  __syncthreads();
  atomicAdd(&feats[img * 256 + threadIdx.x], accum[threadIdx.x] * (1.f / 1024.f));
}

// ---------------------------------------------------------------- GRU
__global__ void k_rkT(const float* __restrict__ rk, bf16* __restrict__ rkT){
  int j = blockIdx.x; int c = threadIdx.x;
  rkT[(size_t)j * 256 + c] = (bf16)rk[(size_t)c * 768 + j];
}

__global__ __launch_bounds__(1024)
void k_gru(const float* __restrict__ mx, const bf16* __restrict__ rkT,
           const float* __restrict__ gb, float* __restrict__ outs,
           float* __restrict__ state){
  __shared__ float h[32][256];
  int tid = threadIdx.x;
  for (int i = tid; i < 8192; i += 1024) ((float*)h)[i] = 0.f;
  __syncthreads();
  const float* b1 = gb + 768;
  for (int t = 0; t < 8; t++){
    float hn[8];
    #pragma unroll
    for (int q = 0; q < 8; q++){
      int id = tid + q * 1024;
      int b_ = id >> 8; int u = id & 255;
      const float* hb = h[b_];
      float mz = 0.f, mr = 0.f, mh = 0.f;
      const bf16* rz = rkT + (size_t)u * 256;
      const bf16* rr = rkT + (size_t)(256 + u) * 256;
      const bf16* rh = rkT + (size_t)(512 + u) * 256;
      for (int c = 0; c < 256; c++){
        float hv = hb[c];
        mz = fmaf(hv, (float)rz[c], mz);
        mr = fmaf(hv, (float)rr[c], mr);
        mh = fmaf(hv, (float)rh[c], mh);
      }
      const float* mxr = mx + ((size_t)(b_ * 8 + t)) * 768;   // feats rows are img = b*8+t
      float zg = 1.f / (1.f + __expf(-(mxr[u]       + mz + b1[u])));
      float rg = 1.f / (1.f + __expf(-(mxr[256 + u] + mr + b1[256 + u])));
      float cd = tanhf(mxr[512 + u] + rg * (mh + b1[512 + u]));
      hn[q] = zg * hb[u] + (1.f - zg) * cd;
    }
    __syncthreads();
    #pragma unroll
    for (int q = 0; q < 8; q++){
      int id = tid + q * 1024; int b_ = id >> 8; int u = id & 255;
      h[b_][u] = hn[q];
      outs[((size_t)t * 32 + b_) * 256 + u] = hn[q];
    }
    __syncthreads();
  }
  for (int i = tid; i < 8192; i += 1024) state[i] = ((float*)h)[i];
}

__global__ void k_full(const float* __restrict__ state, const float* __restrict__ outs,
                       bf16* __restrict__ fullb){
  int i = blockIdx.x * 256 + threadIdx.x;     // 32*2304
  if (i >= 32 * 2304) return;
  int b_ = i / 2304; int j = i % 2304;
  float v;
  if (j < 256) v = state[b_ * 256 + j];
  else { int t = (j - 256) >> 8; int c = (j - 256) & 255; v = outs[((size_t)t * 32 + b_) * 256 + c]; }
  fullb[i] = (bf16)v;
}

// ---------------------------------------------------------------- positional encoding (tanh), bf16
__global__ void k_pos(const float* __restrict__ pw, const float* __restrict__ pb,
                      bf16* __restrict__ posb){
  int i = blockIdx.x * 256 + threadIdx.x;     // 400*256
  if (i >= 102400) return;
  int p = i >> 8; int c = i & 255;
  int yy = p / 20, xx = p % 20;
  float fy = -1.f + 2.f * yy / 19.f, fx = -1.f + 2.f * xx / 19.f;
  posb[i] = (bf16)tanhf(fy * pw[c] + fx * pw[256 + c] + pb[c]);
}

// ---------------------------------------------------------------- attention scalar pipeline
__global__ __launch_bounds__(128)
void k_score(const float* __restrict__ qf, const float* __restrict__ qp,
             const float* __restrict__ kf, const float* __restrict__ kp,
             float* __restrict__ score){
  int b_ = blockIdx.y;
  int p  = blockIdx.x * 4 + (threadIdx.x >> 5);
  int lane = threadIdx.x & 31;
  const float* qfb = qf + b_ * 256; const float* kfb = kf + b_ * 256;
  const float* qpp = qp + (size_t)p * 256; const float* kpp = kp + (size_t)p * 256;
  float s = 0.f;
  for (int c = lane; c < 256; c += 32)
    s += (qfb[c] + qpp[c]) * (kfb[c] + kpp[c]);
  for (int d = 16; d; d >>= 1) s += __shfl_down(s, d, 32);
  if (lane == 0) score[b_ * 400 + p] = s;
}

__global__ void k_rowsum(const float* __restrict__ m, float* __restrict__ out){
  int r = blockIdx.x; int lane = threadIdx.x;
  float s = 0.f;
  for (int c = lane; c < 256; c += 32) s += m[(size_t)r * 256 + c];
  for (int d = 16; d; d >>= 1) s += __shfl_down(s, d, 32);
  if (lane == 0) out[r] = s;
}

__global__ __launch_bounds__(512)
void k_softmax_o(const float* __restrict__ score, const float* __restrict__ SF,
                 const float* __restrict__ SP, float* __restrict__ oscal){
  int b_ = blockIdx.x; int tid = threadIdx.x;
  __shared__ float red[512];
  float v = (tid < 400) ? score[b_ * 400 + tid] : -1e30f;
  red[tid] = v; __syncthreads();
  for (int s = 256; s; s >>= 1){ if (tid < s) red[tid] = fmaxf(red[tid], red[tid + s]); __syncthreads(); }
  float mx_ = red[0]; __syncthreads();
  float e = (tid < 400) ? __expf(v - mx_) : 0.f;
  red[tid] = e; __syncthreads();
  for (int s = 256; s; s >>= 1){ if (tid < s) red[tid] += red[tid + s]; __syncthreads(); }
  float inv = 1.f / red[0];
  if (tid < 400) oscal[b_ * 400 + tid] = e * inv * (SF[b_] + SP[tid]);
}

__global__ void k_ut(const float* __restrict__ ow, const float* __restrict__ ob,
                     const float* __restrict__ chw, const float* __restrict__ chb,
                     float* __restrict__ u, float* __restrict__ t0){
  int c2 = threadIdx.x;    // 1 block of 256
  float su = 0.f, st = 0.f;
  for (int c = 0; c < 256; c++){
    float wv = chw[c * 256 + c2];
    su = fmaf(ow[c], wv, su);
    st = fmaf(ob[c], wv, st);
  }
  u[c2] = su; t0[c2] = st + chb[c2];
}

__global__ __launch_bounds__(256)
void k_xc(const float* __restrict__ oscal, const float* __restrict__ u,
          const float* __restrict__ t0, bf16* __restrict__ xcb,
          float* __restrict__ xcsum){
  int row = blockIdx.x;       // b*400+p
  int c = threadIdx.x;
  int b_ = row / 400;
  float v = fmaxf(oscal[row] * u[c] + t0[c], 0.f);
  xcb[(size_t)row * 256 + c] = (bf16)v;
  atomicAdd(&xcsum[b_ * 256 + c], v * (1.f / 400.f));
}

__global__ void k_wts(const float* __restrict__ xcsum, const float* __restrict__ selw,
                      const float* __restrict__ selb, float* __restrict__ wts){
  __shared__ float lg[32][4];
  int tid = threadIdx.x; int b_ = tid >> 2; int n = tid & 3;  // 128 thr
  float s = selb[n];
  for (int c = 0; c < 256; c++) s = fmaf(xcsum[b_ * 256 + c], selw[c * 4 + n], s);
  lg[b_][n] = s; __syncthreads();
  if (n == 0){
    float m = fmaxf(fmaxf(lg[b_][0], lg[b_][1]), fmaxf(lg[b_][2], lg[b_][3]));
    float e0 = __expf(lg[b_][0] - m), e1 = __expf(lg[b_][1] - m);
    float e2 = __expf(lg[b_][2] - m), e3 = __expf(lg[b_][3] - m);
    float inv = 1.f / (e0 + e1 + e2 + e3);
    wts[b_ * 4 + 0] = e0 * inv; wts[b_ * 4 + 1] = e1 * inv;
    wts[b_ * 4 + 2] = e2 * inv; wts[b_ * 4 + 3] = e3 * inv;
  }
}

__global__ void k_chosen(const float* __restrict__ cands, const float* __restrict__ wts,
                         bf16* __restrict__ chb){
  size_t i = (size_t)blockIdx.x * 256 + threadIdx.x;   // 12800*256
  int row = (int)(i >> 8); int b_ = row / 400;
  const float* w4 = wts + b_ * 4;
  const size_t S = (size_t)12800 * 256;
  float v = cands[i] * w4[0] + cands[i + S] * w4[1]
          + cands[i + 2 * S] * w4[2] + cands[i + 3 * S] * w4[3];
  chb[i] = (bf16)v;
}

__global__ void k_dec2(const bf16* __restrict__ y1, const float* __restrict__ w,
                       const float* __restrict__ b, float* __restrict__ out){
  int i = blockIdx.x * 256 + threadIdx.x;    // 12800*10
  if (i >= 128000) return;
  int row = i / 10, k = i % 10;
  const bf16* yr = y1 + (size_t)row * 256;
  float s = b[k];
  for (int c = 0; c < 256; c++) s = fmaf((float)yr[c], w[c * 10 + k], s);
  out[i] = s;
}

// ---------------------------------------------------------------- host orchestration
extern "C" void kernel_launch(void* const* d_in, const int* in_sizes, int n_in,
                              void* d_out, int out_size, void* d_ws, size_t ws_size,
                              hipStream_t stream){
  (void)in_sizes; (void)n_in; (void)out_size; (void)ws_size;
  const float* x_seq  = (const float*)d_in[0];
  const float* enc_w1 = (const float*)d_in[1];
  const float* enc_b1 = (const float*)d_in[2];
  const float* enc_w2 = (const float*)d_in[3];
  const float* enc_b2 = (const float*)d_in[4];
  const float* ln_g   = (const float*)d_in[5];
  const float* ln_b   = (const float*)d_in[6];
  const float* gru_k  = (const float*)d_in[7];
  const float* gru_rk = (const float*)d_in[8];
  const float* gru_b  = (const float*)d_in[9];
  const float* pos_w  = (const float*)d_in[10];
  const float* pos_b  = (const float*)d_in[11];
  const float* q_w    = (const float*)d_in[12];
  const float* q_b    = (const float*)d_in[13];
  const float* k_w    = (const float*)d_in[14];
  const float* k_b    = (const float*)d_in[15];
  const float* v_w    = (const float*)d_in[16];
  const float* v_b    = (const float*)d_in[17];
  const float* o_w    = (const float*)d_in[18];
  const float* o_b    = (const float*)d_in[19];
  const float* ch_in_w= (const float*)d_in[20];
  const float* ch_in_b= (const float*)d_in[21];
  const float* hyp_w  = (const float*)d_in[22];
  const float* hyp_b  = (const float*)d_in[23];
  const float* sel_w  = (const float*)d_in[24];
  const float* sel_b  = (const float*)d_in[25];
  const float* dec_w1 = (const float*)d_in[26];
  const float* dec_b1 = (const float*)d_in[27];
  const float* dec_w2 = (const float*)d_in[28];
  const float* dec_b2 = (const float*)d_in[29];
  float* out = (float*)d_out;

  char* base = (char*)d_ws;
  size_t off = 0;
  auto alloc = [&](size_t bytes)->char*{
    char* p = base + off;
    off = (off + bytes + 255) & ~(size_t)255;
    return p;
  };
  const size_t NE_IMG = (size_t)256 * 1024 * 256;      // 67,108,864
  bf16*  A1     = (bf16*)alloc(NE_IMG * 2);
  bf16*  ACT2   = (bf16*)alloc(NE_IMG * 2);
  bf16*  W2B    = (bf16*)alloc(589824 * 2);            // [9][co][ci]
  float* FEATS  = (float*)alloc(65536 * 4);
  bf16*  FEATSB = (bf16*)alloc(65536 * 2);
  bf16*  GRUKB  = (bf16*)alloc(196608 * 2);            // [768][256]
  float* MX     = (float*)alloc(196608 * 4);
  bf16*  RKT    = (bf16*)alloc(196608 * 2);
  float* OUTS   = (float*)alloc(65536 * 4);
  float* STATE  = (float*)alloc(8192 * 4);
  bf16*  FULLB  = (bf16*)alloc(73728 * 2);
  bf16*  QWT    = (bf16*)alloc((size_t)589824 * 3 * 2);// [256][2304] x3
  bf16*  QWB    = (bf16*)alloc((size_t)65536 * 3 * 2); // [256][256] x3
  bf16*  POSB   = (bf16*)alloc(102400 * 2);
  float* QF     = (float*)alloc(8192 * 4);
  float* KF     = (float*)alloc(8192 * 4);
  float* VF     = (float*)alloc(8192 * 4);
  float* QP     = (float*)alloc(102400 * 4);
  float* KP     = (float*)alloc(102400 * 4);
  float* VP     = (float*)alloc(102400 * 4);
  float* SCORE  = (float*)alloc(12800 * 4);
  float* SFb    = (float*)alloc(32 * 4);
  float* SPp    = (float*)alloc(400 * 4);
  float* OSCAL  = (float*)alloc(12800 * 4);
  float* Uc     = (float*)alloc(256 * 4);
  float* T0c    = (float*)alloc(256 * 4);
  bf16*  XCB    = (bf16*)alloc((size_t)3276800 * 2);
  float* XCSUM  = (float*)alloc(8192 * 4);
  float* WTS    = (float*)alloc(128 * 4);
  bf16*  HYPWB  = (bf16*)alloc(262144 * 2);            // [n][256][256] transposed
  float* CANDS  = (float*)alloc((size_t)13107200 * 4);
  bf16*  CHOSB  = (bf16*)alloc((size_t)3276800 * 2);
  bf16*  DECW1B = (bf16*)alloc(589824 * 2);            // [9][co][ci]
  bf16*  Y1B    = (bf16*)alloc((size_t)3276800 * 2);

  auto cvt = [&](const float* s, bf16* d, int n){
    k_f2bf<<<(n + 255) / 256, 256, 0, stream>>>(s, d, n);
  };
  auto cvtT = [&](const float* s, bf16* d, int K, int N){
    k_f2bf_T<<<(K * N + 255) / 256, 256, 0, stream>>>(s, d, K, N);
  };

  // zero accumulators
  k_zero<<<(65536 + 255) / 256, 256, 0, stream>>>(FEATS, 65536);
  k_zero<<<(8192 + 255) / 256, 256, 0, stream>>>(XCSUM, 8192);

  // weight conversions to bf16 (B operands transposed to [N][K])
  k_wT<<<(589824 + 255) / 256, 256, 0, stream>>>(enc_w2, W2B);
  k_wT<<<(589824 + 255) / 256, 256, 0, stream>>>(dec_w1, DECW1B);
  cvtT(gru_k, GRUKB, 256, 768);
  cvtT(q_w, QWT, 2304, 256);                 // top block [2304,256] -> [256,2304]
  cvtT(k_w, QWT + 589824, 2304, 256);
  cvtT(v_w, QWT + 2 * 589824, 2304, 256);
  cvtT(q_w + (size_t)2304 * 256, QWB, 256, 256);
  cvtT(k_w + (size_t)2304 * 256, QWB + 65536, 256, 256);
  cvtT(v_w + (size_t)2304 * 256, QWB + 2 * 65536, 256, 256);
  for (int n = 0; n < 4; n++)
    cvtT(hyp_w + (size_t)n * 65536, HYPWB + (size_t)n * 65536, 256, 256);

  // encoder
  k_conv1<<<262144, 256, 0, stream>>>(x_seq, enc_w1, enc_b1, A1);
  k_conv3_wmma<<<256 * 32, 256, 0, stream>>>(A1, W2B, enc_b2, ACT2, 32, 32);
  k_ln_mean<<<1024, 256, 0, stream>>>(ACT2, ln_g, ln_b, FEATS);
  cvt(FEATS, FEATSB, 65536);

  // GRU: mx = feats @ gru_k + b0 (WMMA), then recurrence
  k_gemm_bf16<<<dim3(12, 4), 128, 0, stream>>>(FEATSB, GRUKB, gru_b, MX, 256, 768, 256, 0);
  k_rkT<<<768, 256, 0, stream>>>(gru_rk, RKT);
  k_gru<<<1, 1024, 0, stream>>>(MX, RKT, gru_b, OUTS, STATE);
  k_full<<<288, 256, 0, stream>>>(STATE, OUTS, FULLB);

  // positional encoding + factored q/k/v
  k_pos<<<400, 256, 0, stream>>>(pos_w, pos_b, POSB);
  k_gemm_bf16<<<dim3(4, 1), 128, 0, stream>>>(FULLB, QWT,              q_b, QF, 32, 256, 2304, 0);
  k_gemm_bf16<<<dim3(4, 1), 128, 0, stream>>>(FULLB, QWT + 589824,     k_b, KF, 32, 256, 2304, 0);
  k_gemm_bf16<<<dim3(4, 1), 128, 0, stream>>>(FULLB, QWT + 2 * 589824, v_b, VF, 32, 256, 2304, 0);
  k_gemm_bf16<<<dim3(4, 7), 128, 0, stream>>>(POSB, QWB,              nullptr, QP, 400, 256, 256, 0);
  k_gemm_bf16<<<dim3(4, 7), 128, 0, stream>>>(POSB, QWB + 65536,      nullptr, KP, 400, 256, 256, 0);
  k_gemm_bf16<<<dim3(4, 7), 128, 0, stream>>>(POSB, QWB + 2 * 65536,  nullptr, VP, 400, 256, 256, 0);

  // attention scalar path
  k_score<<<dim3(100, 32), 128, 0, stream>>>(QF, QP, KF, KP, SCORE);
  k_rowsum<<<32, 32, 0, stream>>>(VF, SFb);
  k_rowsum<<<400, 32, 0, stream>>>(VP, SPp);
  k_softmax_o<<<32, 512, 0, stream>>>(SCORE, SFb, SPp, OSCAL);

  // rank-1 collapse of o@o_w -> ch_in
  k_ut<<<1, 256, 0, stream>>>(o_w, o_b, ch_in_w, ch_in_b, Uc, T0c);
  k_xc<<<12800, 256, 0, stream>>>(OSCAL, Uc, T0c, XCB, XCSUM);
  k_wts<<<1, 128, 0, stream>>>(XCSUM, sel_w, sel_b, WTS);

  // 4 hypothesis GEMMs (WMMA) + weighted combine
  for (int n = 0; n < 4; n++)
    k_gemm_bf16<<<dim3(4, 200), 128, 0, stream>>>(XCB, HYPWB + (size_t)n * 65536,
                                                  hyp_b + n * 256,
                                                  CANDS + (size_t)n * 3276800,
                                                  12800, 256, 256, 1);
  k_chosen<<<12800, 256, 0, stream>>>(CANDS, WTS, CHOSB);

  // decoder
  k_conv3_wmma<<<32 * 20, 256, 0, stream>>>(CHOSB, DECW1B, dec_b1, Y1B, 20, 20);
  k_dec2<<<(128000 + 255) / 256, 256, 0, stream>>>(Y1B, dec_w2, dec_b2, out);
}
